// SwinTransformerBlock_21234318312000
// MI455X (gfx1250) — matrix operand order repaired and checked by reading, not compiled, not run on hardware
//
#include <hip/hip_runtime.h>
#include <hip/hip_bf16.h>
#include <stdint.h>

// ---------------------------------------------------------------------------
// CDNA5 (gfx1250) Swin block. Matmuls via v_wmma_f32_16x16x32_f16; GEMM tile
// staging via global_load_async_to_lds_b128 (ASYNCcnt); attention bias+mask
// pre-fused and injected as the WMMA C operand; softmax reductions via
// ds_swizzle xor mode.
// ---------------------------------------------------------------------------

typedef __attribute__((ext_vector_type(16))) _Float16 v16h;
typedef __attribute__((ext_vector_type(8)))  _Float16 v8h;
typedef __attribute__((ext_vector_type(8)))  float    v8f;

#define WMMA_F16(a, b, c) \
  __builtin_amdgcn_wmma_f32_16x16x32_f16(false, (a), false, (b), (short)0, (c), false, false)

// A-fragment (16x32 f16, MxK): lane L holds row M=L&15; K = 16*(L>>4)..+15
// -> one 16-half load.
__device__ __forceinline__ v16h lds_a_frag(const _Float16* base, int ld, int lane) {
  const int r = lane & 15, h = lane >> 4;
  return *(const v16h*)(base + r * ld + 16 * h);
}

// B-fragment (32x16 f16, KxN) loaded from the TRANSPOSED (N-major, N x K)
// storage: lane L holds col N=L&15, K = 16*(L>>4)..+15 -> one 16-half load.
__device__ __forceinline__ v16h lds_b_frag(const _Float16* base, int ld, int lane) {
  const int c = lane & 15, h = lane >> 4;
  return *(const v16h*)(base + c * ld + 16 * h);
}

// memory -> LDS direct copy, 16B per lane, tracked by ASYNCcnt.
__device__ __forceinline__ void async_b128(uint32_t lds_off, const void* gptr) {
  asm volatile("global_load_async_to_lds_b128 %0, %1, off"
               :: "v"(lds_off), "v"((uint64_t)(uintptr_t)gptr) : "memory");
}
__device__ __forceinline__ void wait_async0() {
  asm volatile("s_wait_asynccnt 0" ::: "memory");
}

// ds_swizzle xor within 16-lane groups (and=0x1f, or=0, xor=M): one DS op,
// no bpermute index math.
template <int M>
__device__ __forceinline__ float swz_xor(float x) {
  return __int_as_float(__builtin_amdgcn_ds_swizzle(__float_as_int(x), 0x7C00 | M));
}
__device__ __forceinline__ float red_max16(float v) {
  v = fmaxf(v, swz_xor<8>(v)); v = fmaxf(v, swz_xor<4>(v));
  v = fmaxf(v, swz_xor<2>(v)); v = fmaxf(v, swz_xor<1>(v));
  return v;
}
__device__ __forceinline__ float red_sum16(float v) {
  v += swz_xor<8>(v); v += swz_xor<4>(v);
  v += swz_xor<2>(v); v += swz_xor<1>(v);
  return v;
}

// ---------------------------------------------------------------------------
__global__ void cvt_f16_kernel(const float* __restrict__ in, _Float16* __restrict__ out, int n) {
  int i = blockIdx.x * 256 + threadIdx.x;
  if (i < n) out[i] = (_Float16)in[i];
}

__global__ void zero_u32_kernel(uint32_t* __restrict__ p, int n) {
  int i = blockIdx.x * 256 + threadIdx.x;
  if (i < n) p[i] = 0u;
}

// Fused (rel_pos_bias + shift_mask) 64x64 fp32 tile per (window-in-image, head),
// pad key columns pre-masked with -1e30. Used as WMMA C operand in attention.
__global__ __launch_bounds__(256)
void build_cbias_kernel(const float* __restrict__ relTab, const float* __restrict__ mask,
                        float* __restrict__ cb) {
  const int head = blockIdx.x, wi = blockIdx.y;
  float* dst = cb + ((size_t)wi * 12 + head) * 4096;
  const float* wmask = mask + (size_t)wi * 2401;
  for (int i = threadIdx.x; i < 4096; i += 256) {
    const int m = i >> 6, n = i & 63;
    float v;
    if (m < 49 && n < 49) {
      const int iq = m / 7, jq = m % 7, ik = n / 7, jk = n % 7;
      v = relTab[((iq - ik + 6) * 13 + (jq - jk + 6)) * 12 + head] + wmask[m * 49 + n];
    } else {
      v = (n >= 49) ? -1e30f : 0.0f;   // mask pad keys; pad query rows harmless
    }
    dst[i] = v;
  }
}

// ---------------------------------------------------------------------------
// LayerNorm over C=384, one token per block (128 threads, 3 ch/thread).
// WINDOWED=true also performs the cyclic shift + 7x7 window partition and
// writes into the 64-token-padded window layout (pad rows pre-zeroed).
template <bool WINDOWED>
__global__ __launch_bounds__(128)
void ln_kernel(const float* __restrict__ x, const float* __restrict__ gw,
               const float* __restrict__ gb, _Float16* __restrict__ out) {
  __shared__ float red[4];
  const int tid = threadIdx.x;
  const int gid = blockIdx.x;
  const float* src;
  _Float16* dst;
  if (WINDOWED) {
    const int win = gid / 49, nn = gid - win * 49;
    const int b = win >> 6, wi = win & 63;
    const int hs = (wi >> 3) * 7 + nn / 7;
    const int wsx = (wi & 7) * 7 + nn % 7;
    int hh = hs + 3; if (hh >= 56) hh -= 56;   // roll(-3): xs[hs] = xn[(hs+3)%56]
    int ww = wsx + 3; if (ww >= 56) ww -= 56;
    src = x + ((size_t)(b * 56 + hh) * 56 + ww) * 384;
    dst = out + ((size_t)win * 64 + nn) * 384;
  } else {
    src = x + (size_t)gid * 384;
    dst = out + (size_t)gid * 384;
  }
  float v0 = src[tid], v1 = src[tid + 128], v2 = src[tid + 256];
  float s = v0 + v1 + v2;
  for (int off = 16; off; off >>= 1) s += __shfl_xor(s, off, 32);
  if ((tid & 31) == 0) red[tid >> 5] = s;
  __syncthreads();
  const float mean = (red[0] + red[1] + red[2] + red[3]) * (1.0f / 384.0f);
  __syncthreads();
  const float d0 = v0 - mean, d1 = v1 - mean, d2 = v2 - mean;
  s = d0 * d0 + d1 * d1 + d2 * d2;
  for (int off = 16; off; off >>= 1) s += __shfl_xor(s, off, 32);
  if ((tid & 31) == 0) red[tid >> 5] = s;
  __syncthreads();
  const float var = (red[0] + red[1] + red[2] + red[3]) * (1.0f / 384.0f);
  const float rs = rsqrtf(var + 1e-5f);
  dst[tid]       = (_Float16)(d0 * rs * gw[tid]       + gb[tid]);
  dst[tid + 128] = (_Float16)(d1 * rs * gw[tid + 128] + gb[tid + 128]);
  dst[tid + 256] = (_Float16)(d2 * rs * gw[tid + 256] + gb[tid + 256]);
}

// ---------------------------------------------------------------------------
// GEMM:  out[M,N] = A[M,K](f16 row-major) * W[N,K](f16 row-major, i.e. x@W^T)
// Block tile 128x128, BK=32, 8 waves (4x2); each wave 32x64 = 2x4 WMMA tiles
// (8 WMMAs / k-step). Tiles staged with async memory->LDS copies, dbl-buffered.
// EPI: 0=bias->f16, 1=bias+GELU->f16, 2=bias+window-reverse+shift+residual->f32,
//      3=bias+residual->f32.
template <int EPI>
__global__ __launch_bounds__(256)
void gemm_f16_kernel(const _Float16* __restrict__ A, const _Float16* __restrict__ W,
                     const float* __restrict__ bias, const float* __restrict__ resid,
                     void* __restrict__ out, int M, int N, int K) {
  __shared__ _Float16 sA[2][128 * 32];
  __shared__ _Float16 sB[2][128 * 32];
  const int tid = threadIdx.x, lane = tid & 31, wid = tid >> 5;
  const int wm = wid & 3, wn = wid >> 2;
  const int blockM = blockIdx.y * 128;
  const int blockN = blockIdx.x * 128;

  // staging: 128 rows x 32 halfs per tile; thread -> (row, 16-half segment)
  const int row = tid >> 1, seg = tid & 1;
  const _Float16* aPtr = A + (size_t)(blockM + row) * K + 16 * seg;
  const _Float16* bPtr = W + (size_t)(blockN + row) * K + 16 * seg;
  const uint32_t laBase = (uint32_t)(uintptr_t)&sA[0][row * 32 + 16 * seg];
  const uint32_t lbBase = (uint32_t)(uintptr_t)&sB[0][row * 32 + 16 * seg];
  const uint32_t bufStride = (uint32_t)(128 * 32 * 2);  // bytes per LDS buffer

  auto stage = [&](int kb, int buf) {
    async_b128(laBase + (uint32_t)buf * bufStride, aPtr + kb);
    async_b128(laBase + (uint32_t)buf * bufStride + 16, aPtr + kb + 8);
    async_b128(lbBase + (uint32_t)buf * bufStride, bPtr + kb);
    async_b128(lbBase + (uint32_t)buf * bufStride + 16, bPtr + kb + 8);
  };

  stage(0, 0);
  wait_async0();
  __syncthreads();

  v8f acc[2][4] = {};
  const int nk = K >> 5;
  for (int kk = 0; kk < nk; ++kk) {
    const int buf = kk & 1;
    if (kk + 1 < nk) {
      stage((kk + 1) << 5, buf ^ 1);
      if (kk + 2 < nk) __builtin_prefetch(aPtr + ((kk + 2) << 5), 0, 1);
    }
    v16h af[2], bf[4];
    for (int a = 0; a < 2; ++a)
      af[a] = lds_a_frag(&sA[buf][(32 * wm + 16 * a) * 32], 32, lane);
    for (int bt = 0; bt < 4; ++bt)
      bf[bt] = lds_b_frag(&sB[buf][(64 * wn + 16 * bt) * 32], 32, lane);
    for (int a = 0; a < 2; ++a)
      for (int bt = 0; bt < 4; ++bt)
        acc[a][bt] = WMMA_F16(af[a], bf[bt], acc[a][bt]);
    wait_async0();
    __syncthreads();
  }

  // C/D layout: lane L -> col N=L&15; VGPR j -> row M = j + 8*(L>>4).
  const int col = lane & 15, h = lane >> 4;
  for (int a = 0; a < 2; ++a)
    for (int bt = 0; bt < 4; ++bt) {
      v8f v = acc[a][bt];
      for (int j = 0; j < 8; ++j) {
        const int m = blockM + 32 * wm + 16 * a + 8 * h + j;
        const int n = blockN + 64 * wn + 16 * bt + col;
        float val = v[j] + bias[n];
        if (EPI == 0) {
          ((_Float16*)out)[(size_t)m * N + n] = (_Float16)val;
        } else if (EPI == 1) {
          const float g = 0.5f * val * (1.0f + erff(val * 0.70710678118654752f));
          ((_Float16*)out)[(size_t)m * N + n] = (_Float16)g;
        } else if (EPI == 2) {
          const int win = m >> 6, nn = m & 63;
          if (nn < 49) {  // drop pad rows; window-reverse + roll(+3) + residual
            const int bimg = win >> 6, wi = win & 63;
            const int hs = (wi >> 3) * 7 + nn / 7;
            const int wsx = (wi & 7) * 7 + nn % 7;
            int hh = hs + 3; if (hh >= 56) hh -= 56;
            int ww = wsx + 3; if (ww >= 56) ww -= 56;
            const size_t idx = ((size_t)(bimg * 56 + hh) * 56 + ww) * 384 + n;
            ((float*)out)[idx] = resid[idx] + val;
          }
        } else {
          const size_t idx = (size_t)m * N + n;
          ((float*)out)[idx] = resid[idx] + val;
        }
      }
    }
}

// ---------------------------------------------------------------------------
// Attention: one block per (window, head). 4 waves, wave w owns score rows
// 16w..16w+15 of the 64x64 (49 valid) score matrix. Scores = (q*scale)@k^T + C
// where C is the prebuilt fused bias+mask tile, injected as the WMMA C operand.
__global__ __launch_bounds__(128)
void attn_kernel(const _Float16* __restrict__ qkv, const float* __restrict__ cbias,
                 _Float16* __restrict__ out) {
  __shared__ _Float16 qs[64 * 32];
  __shared__ _Float16 ks[64 * 32];
  __shared__ _Float16 vt[32 * 64];   // V transposed: [hd][token]
  __shared__ _Float16 ps[64 * 64];   // probabilities
  const int head = blockIdx.x;
  const int win = blockIdx.y;
  const int tid = threadIdx.x, lane = tid & 31, wid = tid >> 5;

  for (int i = tid; i < 64 * 32; i += 128) {
    qs[i] = (_Float16)0.f; ks[i] = (_Float16)0.f; vt[i] = (_Float16)0.f;
  }
  __syncthreads();
  const float scale = 0.17677669529663687f;  // 32^-0.5, folded into q
  const _Float16* base = qkv + (size_t)win * 64 * 1152 + head * 32;
  if (tid < 98) {  // 49 rows x two 16-half chunks, vectorized b128 loads
    const int r = tid >> 1, sg = tid & 1;
    const _Float16* rp = base + (size_t)r * 1152 + sg * 16;
    v16h qv = *(const v16h*)rp;
    v16h kv = *(const v16h*)(rp + 384);
    v16h vv = *(const v16h*)(rp + 768);
    for (int e = 0; e < 16; ++e)
      qs[r * 32 + sg * 16 + e] = (_Float16)((float)qv[e] * scale);
    *(v16h*)&ks[r * 32 + sg * 16] = kv;
    for (int e = 0; e < 16; ++e)
      vt[(sg * 16 + e) * 64 + r] = vv[e];
  }
  __syncthreads();

  const int r0 = wid * 16;
  const int col = lane & 15, h = lane >> 4;
  const float* cbt = cbias + ((size_t)(win & 63) * 12 + head) * 4096;

  // S = (q*scale) @ k^T + cbias : one WMMA (K=32 = head dim) per 16x16 tile.
  v16h aq = lds_a_frag(&qs[r0 * 32], 32, lane);
  float sc[4][8];
  for (int tn = 0; tn < 4; ++tn) {
    v16h bk = lds_b_frag(&ks[(tn * 16) * 32], 32, lane);
    v8f cin;
    for (int j = 0; j < 8; ++j)
      cin[j] = cbt[(r0 + 8 * h + j) * 64 + tn * 16 + col];
    v8f a = WMMA_F16(aq, bk, cin);
    for (int j = 0; j < 8; ++j) sc[tn][j] = a[j];
  }

  // softmax across the 64 columns of each row (16 lanes per half-wave x 4 tiles)
  for (int j = 0; j < 8; ++j) {
    const int m = r0 + 8 * h + j;
    float mx = red_max16(fmaxf(fmaxf(sc[0][j], sc[1][j]), fmaxf(sc[2][j], sc[3][j])));
    float sum = 0.f;
    for (int tn = 0; tn < 4; ++tn) {
      const float e = __expf(sc[tn][j] - mx);
      sc[tn][j] = e; sum += e;
    }
    sum = red_sum16(sum);
    const float inv = 1.0f / sum;
    for (int tn = 0; tn < 4; ++tn)
      ps[m * 64 + tn * 16 + col] = (_Float16)(sc[tn][j] * inv);
  }
  __syncthreads();

  // O = P @ V : K=64 -> 2 WMMA k-steps; B-frags from transposed V (contiguous).
  v16h pa0 = lds_a_frag(&ps[r0 * 64], 64, lane);
  v16h pa1 = lds_a_frag(&ps[r0 * 64 + 32], 64, lane);
  for (int tn = 0; tn < 2; ++tn) {
    v8f acc = {};
    acc = WMMA_F16(pa0, lds_b_frag(&vt[(tn * 16) * 64], 64, lane), acc);
    acc = WMMA_F16(pa1, lds_b_frag(&vt[(tn * 16) * 64 + 32], 64, lane), acc);
    for (int j = 0; j < 8; ++j) {
      const int m = r0 + 8 * h + j;
      out[((size_t)win * 64 + m) * 384 + head * 32 + tn * 16 + col] = (_Float16)acc[j];
    }
  }
}

// ---------------------------------------------------------------------------
extern "C" void kernel_launch(void* const* d_in, const int* in_sizes, int n_in,
                              void* d_out, int out_size, void* d_ws, size_t ws_size,
                              hipStream_t stream) {
  (void)in_sizes; (void)n_in; (void)out_size; (void)ws_size;
  const float* x     = (const float*)d_in[0];
  const float* amask = (const float*)d_in[1];
  const float* n1w   = (const float*)d_in[2];
  const float* n1b   = (const float*)d_in[3];
  const float* qkvW  = (const float*)d_in[4];
  const float* qkvB  = (const float*)d_in[5];
  const float* relT  = (const float*)d_in[6];
  const float* projW = (const float*)d_in[7];
  const float* projB = (const float*)d_in[8];
  const float* n2w   = (const float*)d_in[9];
  const float* n2b   = (const float*)d_in[10];
  const float* fc1W  = (const float*)d_in[11];
  const float* fc1B  = (const float*)d_in[12];
  const float* fc2W  = (const float*)d_in[13];
  const float* fc2B  = (const float*)d_in[14];

  // Workspace layout (aliased regions noted); ~298 MB total.
  char* p = (char*)d_ws;
  _Float16* wq = (_Float16*)p; p += (size_t)1152 * 384 * 2;
  _Float16* wp = (_Float16*)p; p += (size_t)384 * 384 * 2;
  _Float16* w1 = (_Float16*)p; p += (size_t)1536 * 384 * 2;
  _Float16* w2 = (_Float16*)p; p += (size_t)384 * 1536 * 2;
  float*    cb = (float*)p;    p += (size_t)64 * 12 * 4096 * 4;  // fused bias+mask
  _Float16* xw = (_Float16*)p; p += (size_t)65536 * 384 * 2;     // also attn_out, h_ln
  _Float16* qk = (_Float16*)p; p += (size_t)50176 * 1536 * 2;    // qkv, then gelu
  float*    x1 = (float*)p;    p += (size_t)50176 * 384 * 4;

  // weights fp32 -> f16 (kept in (N,K) layout == x @ W^T with no transpose)
  cvt_f16_kernel<<<(1152 * 384 + 255) / 256, 256, 0, stream>>>(qkvW, wq, 1152 * 384);
  cvt_f16_kernel<<<(384 * 384 + 255) / 256, 256, 0, stream>>>(projW, wp, 384 * 384);
  cvt_f16_kernel<<<(1536 * 384 + 255) / 256, 256, 0, stream>>>(fc1W, w1, 1536 * 384);
  cvt_f16_kernel<<<(384 * 1536 + 255) / 256, 256, 0, stream>>>(fc2W, w2, 384 * 1536);

  // fused rel-pos-bias + shift-mask tiles (one per window-in-image x head)
  build_cbias_kernel<<<dim3(12, 64), 256, 0, stream>>>(relT, amask, cb);

  // zero padded-window buffer (rows 49..63 of each window must be 0)
  const int zn = 65536 * 384 / 2;
  zero_u32_kernel<<<(zn + 255) / 256, 256, 0, stream>>>((uint32_t*)xw, zn);

  // LN1 + cyclic shift + window partition -> xw (1024 win x 64 tok x 384) f16
  ln_kernel<true><<<1024 * 49, 128, 0, stream>>>(x, n1w, n1b, xw);

  // QKV: (65536 x 384) @ (384 x 1152)
  gemm_f16_kernel<0><<<dim3(1152 / 128, 65536 / 128), 256, 0, stream>>>(
      xw, wq, qkvB, nullptr, qk, 65536, 1152, 384);

  // Attention per (window, head); writes attn_out (aliases xw)
  attn_kernel<<<dim3(12, 1024), 128, 0, stream>>>(qk, cb, xw);

  // Proj + window reverse + reverse shift + residual(x) -> x1 fp32
  gemm_f16_kernel<2><<<dim3(384 / 128, 65536 / 128), 256, 0, stream>>>(
      xw, wp, projB, x, x1, 65536, 384, 384);

  // LN2 -> h_ln (aliases xw; only first 50176 rows used)
  ln_kernel<false><<<50176, 128, 0, stream>>>(x1, n2w, n2b, xw);

  // FC1 + GELU: (50176 x 384) @ (384 x 1536) -> gelu (aliases qk)
  gemm_f16_kernel<1><<<dim3(1536 / 128, 50176 / 128), 256, 0, stream>>>(
      xw, w1, fc1B, nullptr, qk, 50176, 1536, 384);

  // FC2 + residual(x1): (50176 x 1536) @ (1536 x 384) -> d_out fp32
  gemm_f16_kernel<3><<<dim3(384 / 128, 50176 / 128), 256, 0, stream>>>(
      qk, w2, fc2B, x1, d_out, 50176, 384, 1536);
}
